// EnsembleRSSM_14628658610828
// MI455X (gfx1250) — matrix operand order repaired and checked by reading, not compiled, not run on hardware
//
#include <hip/hip_runtime.h>
#include <hip/hip_bf16.h>
#include <math.h>

// ---------- types for CDNA5 WMMA ----------
typedef __bf16 bf16;
typedef __attribute__((ext_vector_type(16))) __bf16 v16bf;
typedef __attribute__((ext_vector_type(8)))  float  v8f;
typedef __attribute__((ext_vector_type(4)))  unsigned int u32x4;

union Frag { v16bf v; u32x4 q[2]; };

// Problem constants
#define BB 256
#define TT 64
#define SS 32
#define DD 1024
#define HH 1024
#define EE 1536
#define AA 6
#define CC 2240   // 2*(3*S + D)

// ---------------------------------------------------------------------------
// Generic bf16 WMMA GEMM:  C[M,N] = A[M,K] (row-major bf16) * Bt[N,K]^T
// Bt is the transposed weight (row n holds column n of W), so each lane's
// B fragment is two 16-byte K runs (ISA B 32x16 lane map). Block = 256
// threads = 8 waves in a 4x2 grid; each wave computes a 32x32 tile:
// 2 A-frags x 2 B-frags -> 4 v_wmma per k-step (2 b128 loads per wmma).
// Block tile: 128(M) x 64(N). Requires M % 128 == 0, N % 64 == 0, K % 32 == 0.
// mode 0: store f32 (no bias)          -> Cf
// mode 1: +bias, ELU, store bf16       -> Cb
// mode 2: +bias, store f32             -> Cf
// If eidx != nullptr, Bt/bias are offset by eidx[t]*stride (ensemble select).
// ---------------------------------------------------------------------------
__global__ __launch_bounds__(256) void gemm_bf16_wmma(
    const bf16* __restrict__ A, const bf16* __restrict__ Bt,
    int M, int N, int K,
    float* __restrict__ Cf, bf16* __restrict__ Cb,
    const float* __restrict__ bias,
    const int* __restrict__ eidx, int t,
    long bStride, long biasStride, int mode)
{
    if (eidx) {
        int id = eidx[t];
        Bt += (size_t)id * (size_t)bStride;
        if (bias) bias += (size_t)id * (size_t)biasStride;
    }
    const int lane = threadIdx.x & 31;
    const int wave = threadIdx.x >> 5;
    const int wm = wave & 3, wn = wave >> 2;             // 4 x 2 wave grid
    const int rowBase = blockIdx.y * 128 + wm * 32;
    const int colBase = blockIdx.x * 64 + wn * 32;
    const int lh = lane & 15, hh = lane >> 4;

    // A 16x32 bf16 lane map: lane<16 holds K 0..7 & 16..23; lane>=16: 8..15 & 24..31
    const bf16* aPtr0 = A + (size_t)(rowBase + lh) * K;
    const bf16* aPtr1 = A + (size_t)(rowBase + 16 + lh) * K;
    // B 32x16 lane map: lane<16 holds K 0..7 & 16..23; lane>=16: 8..15 & 24..31
    const bf16* bPtr0 = Bt + (size_t)(colBase + lh) * K;
    const bf16* bPtr1 = Bt + (size_t)(colBase + 16 + lh) * K;

    v8f acc00 = {0.f,0.f,0.f,0.f,0.f,0.f,0.f,0.f};
    v8f acc01 = acc00, acc10 = acc00, acc11 = acc00;

    for (int k0 = 0; k0 < K; k0 += 32) {
        Frag fa0, fa1, fb0, fb1;
        fa0.q[0] = *(const u32x4*)(aPtr0 + k0 + hh * 8);
        fa0.q[1] = *(const u32x4*)(aPtr0 + k0 + 16 + hh * 8);
        fa1.q[0] = *(const u32x4*)(aPtr1 + k0 + hh * 8);
        fa1.q[1] = *(const u32x4*)(aPtr1 + k0 + 16 + hh * 8);
        fb0.q[0] = *(const u32x4*)(bPtr0 + k0 + hh * 8);
        fb0.q[1] = *(const u32x4*)(bPtr0 + k0 + 16 + hh * 8);
        fb1.q[0] = *(const u32x4*)(bPtr1 + k0 + hh * 8);
        fb1.q[1] = *(const u32x4*)(bPtr1 + k0 + 16 + hh * 8);
        acc00 = __builtin_amdgcn_wmma_f32_16x16x32_bf16(
                    false, fa0.v, false, fb0.v, (short)0, acc00, false, false);
        acc01 = __builtin_amdgcn_wmma_f32_16x16x32_bf16(
                    false, fa0.v, false, fb1.v, (short)0, acc01, false, false);
        acc10 = __builtin_amdgcn_wmma_f32_16x16x32_bf16(
                    false, fa1.v, false, fb0.v, (short)0, acc10, false, false);
        acc11 = __builtin_amdgcn_wmma_f32_16x16x32_bf16(
                    false, fa1.v, false, fb1.v, (short)0, acc11, false, false);
    }

    // C/D f32 16x16 lane map: col = lane%16, rows r + 8*(lane/16)
    const int n0 = colBase + lh;
    const int n1 = n0 + 16;
    float b0 = 0.f, b1 = 0.f;
    if (mode != 0 && bias) { b0 = bias[n0]; b1 = bias[n1]; }

    #pragma unroll
    for (int r = 0; r < 8; ++r) {
        int mA = rowBase + r + 8 * hh;          // rows of acc0x
        int mB = mA + 16;                       // rows of acc1x
        size_t o00 = (size_t)mA * N + n0, o01 = (size_t)mA * N + n1;
        size_t o10 = (size_t)mB * N + n0, o11 = (size_t)mB * N + n1;
        float f00 = acc00[r], f01 = acc01[r];
        float f10 = acc10[r], f11 = acc11[r];
        if (mode == 0) {
            Cf[o00] = f00; Cf[o01] = f01; Cf[o10] = f10; Cf[o11] = f11;
        } else if (mode == 1) {
            f00 += b0; f01 += b1; f10 += b0; f11 += b1;
            f00 = f00 > 0.f ? f00 : (expf(f00) - 1.f);   // ELU
            f01 = f01 > 0.f ? f01 : (expf(f01) - 1.f);
            f10 = f10 > 0.f ? f10 : (expf(f10) - 1.f);
            f11 = f11 > 0.f ? f11 : (expf(f11) - 1.f);
            Cb[o00] = (bf16)f00; Cb[o01] = (bf16)f01;
            Cb[o10] = (bf16)f10; Cb[o11] = (bf16)f11;
        } else {
            Cf[o00] = f00 + b0; Cf[o01] = f01 + b1;
            Cf[o10] = f10 + b0; Cf[o11] = f11 + b1;
        }
    }
}

// ---------------------------------------------------------------------------
// Per-step prep: apply is_first mask to stoch/action/deter, compute
// x = elu([stoch,a] @ W_img_in + b) (tiny K=38 GEMM, VALU), build bf16
// concat buffer xd = [x | deter_masked], and fill de[:,1024:2560] = bf16(embed).
// ---------------------------------------------------------------------------
__global__ __launch_bounds__(256) void prep_kernel(
    const float* __restrict__ embed, const float* __restrict__ action,
    const unsigned char* __restrict__ isf,
    const float* __restrict__ Wimg, const float* __restrict__ bimg,
    const float* __restrict__ stoch, float* __restrict__ deter,
    bf16* __restrict__ xd, bf16* __restrict__ de, int t)
{
    int tid = blockIdx.x * blockDim.x + threadIdx.x;
    int b = tid / 3584;
    int c = tid % 3584;
    if (b >= BB) return;
    float m = isf[(size_t)b * TT + t] ? 0.f : 1.f;

    if (c < HH) {
        float acc = bimg[c];
        #pragma unroll 4
        for (int k = 0; k < SS; ++k)
            acc += stoch[b * SS + k] * m * Wimg[k * HH + c];
        const float* arow = action + ((size_t)b * TT + t) * AA;
        #pragma unroll
        for (int k = 0; k < AA; ++k)
            acc += arow[k] * m * Wimg[(SS + k) * HH + c];
        float x = acc > 0.f ? acc : (expf(acc) - 1.f);
        xd[(size_t)b * 2048 + c] = (bf16)x;
    } else if (c < 2048) {
        int j = c - HH;
        float dm = deter[(size_t)b * DD + j] * m;
        deter[(size_t)b * DD + j] = dm;
        xd[(size_t)b * 2048 + c] = (bf16)dm;
    } else {
        int k = c - 2048;
        de[(size_t)b * 2560 + DD + k] =
            (bf16)embed[((size_t)b * TT + t) * EE + k];
    }
}

// ---------------------------------------------------------------------------
// LayerNorm over full 3072 row + GRU gate math. One block per batch row.
// Writes new deter (f32 + bf16), de[:,0:1024], and deter output slices.
// ---------------------------------------------------------------------------
__global__ __launch_bounds__(256) void ln_gru_kernel(
    const float* __restrict__ parts, const float* __restrict__ g,
    const float* __restrict__ bl,
    float* __restrict__ deter, bf16* __restrict__ deter_bf,
    bf16* __restrict__ de, float* __restrict__ out, int t)
{
    int b = blockIdx.x;
    const float* row = parts + (size_t)b * 3072;
    __shared__ float s1[256], s2[256];
    float sum = 0.f, sq = 0.f;
    for (int i = threadIdx.x; i < 3072; i += 256) {
        float v = row[i]; sum += v; sq += v * v;
    }
    s1[threadIdx.x] = sum; s2[threadIdx.x] = sq;
    __syncthreads();
    for (int off = 128; off > 0; off >>= 1) {
        if ((int)threadIdx.x < off) {
            s1[threadIdx.x] += s1[threadIdx.x + off];
            s2[threadIdx.x] += s2[threadIdx.x + off];
        }
        __syncthreads();
    }
    float mean = s1[0] * (1.f / 3072.f);
    float var  = s2[0] * (1.f / 3072.f) - mean * mean;
    float rstd = rsqrtf(var + 1e-5f);

    size_t obase = ((size_t)b * TT + t) * CC;
    for (int j = threadIdx.x; j < DD; j += 256) {
        float p0 = (row[j]        - mean) * rstd * g[j]        + bl[j];
        float p1 = (row[1024 + j] - mean) * rstd * g[1024 + j] + bl[1024 + j];
        float p2 = (row[2048 + j] - mean) * rstd * g[2048 + j] + bl[2048 + j];
        float reset = 1.f / (1.f + expf(-p0));
        float cand  = tanhf(reset * p1);
        float upd   = 1.f / (1.f + expf(-(p2 - 1.f)));   // update_bias = -1
        float dn = upd * cand + (1.f - upd) * deter[(size_t)b * DD + j];
        deter[(size_t)b * DD + j]    = dn;
        deter_bf[(size_t)b * DD + j] = (bf16)dn;
        de[(size_t)b * 2560 + j]     = (bf16)dn;
        out[obase + 96 + j]   = dn;    // post deter
        out[obase + 1216 + j] = dn;    // prior deter
    }
}

// ---------------------------------------------------------------------------
// Distribution stats: mean, softplus(std)+MIN_STD, rsample; write out slices.
// outOff=1120 -> prior; outOff=0 -> posterior (also writes stoch carry).
// ---------------------------------------------------------------------------
__global__ __launch_bounds__(256) void stats_kernel(
    const float* __restrict__ raw, const float* __restrict__ noise,
    float* __restrict__ out, float* __restrict__ stoch, int t, int outOff)
{
    int tid = blockIdx.x * blockDim.x + threadIdx.x;
    if (tid >= BB * SS) return;
    int b = tid >> 5, s = tid & 31;
    float mean = raw[(size_t)b * 64 + s];
    float sr   = raw[(size_t)b * 64 + 32 + s];
    float std  = fmaxf(sr, 0.f) + log1pf(expf(-fabsf(sr))) + 0.1f; // softplus+MIN_STD
    float nz   = noise[((size_t)t * BB + b) * SS + s];             // [T,B,S]
    float st   = mean + std * nz;
    size_t obase = ((size_t)b * TT + t) * CC + outOff;
    out[obase + s]      = mean;
    out[obase + 32 + s] = std;
    out[obase + 64 + s] = st;
    if (stoch) stoch[(size_t)b * SS + s] = st;
}

// One-time: f32 [nmat,rows,cols] -> bf16 [nmat,cols,rows] (transposed weights)
__global__ __launch_bounds__(256) void convT_kernel(
    const float* __restrict__ src, bf16* __restrict__ dst,
    int rows, int cols, int nmat)
{
    size_t tid = (size_t)blockIdx.x * blockDim.x + threadIdx.x;
    size_t per = (size_t)rows * cols;
    if (tid >= per * (size_t)nmat) return;
    size_t mat = tid / per, rem = tid % per;
    int r = (int)(rem / cols), c = (int)(rem % cols);
    dst[mat * per + (size_t)c * rows + r] = (bf16)src[tid];
}

__global__ __launch_bounds__(256) void zero_kernel(float* __restrict__ stoch,
                                                   float* __restrict__ deter)
{
    int tid = blockIdx.x * blockDim.x + threadIdx.x;
    if (tid < BB * SS) stoch[tid] = 0.f;
    if (tid < BB * DD) deter[tid] = 0.f;
}

// ---------------------------------------------------------------------------
extern "C" void kernel_launch(void* const* d_in, const int* in_sizes, int n_in,
                              void* d_out, int out_size, void* d_ws, size_t ws_size,
                              hipStream_t stream)
{
    const float* embed  = (const float*)d_in[0];
    const float* action = (const float*)d_in[1];
    const unsigned char* isf = (const unsigned char*)d_in[2];
    const float* npri   = (const float*)d_in[3];
    const float* npost  = (const float*)d_in[4];
    const int*   eidx   = (const int*)d_in[5];
    const float* Wimg   = (const float*)d_in[6];
    const float* bimg   = (const float*)d_in[7];
    const float* Wgru   = (const float*)d_in[8];
    const float* ggru   = (const float*)d_in[9];
    const float* blgru  = (const float*)d_in[10];
    const float* Weo    = (const float*)d_in[11];
    const float* beo    = (const float*)d_in[12];
    const float* Wed    = (const float*)d_in[13];
    const float* bed    = (const float*)d_in[14];
    const float* Woo    = (const float*)d_in[15];
    const float* boo    = (const float*)d_in[16];
    const float* Wod    = (const float*)d_in[17];
    const float* bod    = (const float*)d_in[18];
    float* out = (float*)d_out;
    (void)in_sizes; (void)n_in; (void)out_size; (void)ws_size;

    // ---- carve workspace (~38 MB) ----
    char* p = (char*)d_ws;
    auto carve = [&](size_t bytes) -> void* {
        void* r = (void*)p; p += (bytes + 255) & ~(size_t)255; return r;
    };
    bf16* WgT  = (bf16*)carve((size_t)3072 * 2048 * 2);
    bf16* WeoT = (bf16*)carve((size_t)5 * 1024 * 1024 * 2);
    bf16* WedT = (bf16*)carve((size_t)5 * 64 * 1024 * 2);
    bf16* WooT = (bf16*)carve((size_t)1024 * 2560 * 2);
    bf16* WodT = (bf16*)carve((size_t)64 * 1024 * 2);
    bf16* xd     = (bf16*)carve((size_t)BB * 2048 * 2);
    bf16* de     = (bf16*)carve((size_t)BB * 2560 * 2);
    bf16* deterB = (bf16*)carve((size_t)BB * DD * 2);
    bf16* xoB    = (bf16*)carve((size_t)BB * HH * 2);
    bf16* xpB    = (bf16*)carve((size_t)BB * HH * 2);
    float* parts  = (float*)carve((size_t)BB * 3072 * 4);
    float* deterF = (float*)carve((size_t)BB * DD * 4);
    float* stochF = (float*)carve((size_t)BB * SS * 4);
    float* edOut  = (float*)carve((size_t)BB * 64 * 4);
    float* odOut  = (float*)carve((size_t)BB * 64 * 4);

    // ---- prologue: zero carry state; convert+transpose weights to bf16 ----
    zero_kernel<<<(BB * DD + 255) / 256, 256, 0, stream>>>(stochF, deterF);
    convT_kernel<<<((size_t)2048 * 3072 + 255) / 256, 256, 0, stream>>>(Wgru, WgT, 2048, 3072, 1);
    convT_kernel<<<((size_t)5 * 1024 * 1024 + 255) / 256, 256, 0, stream>>>(Weo, WeoT, 1024, 1024, 5);
    convT_kernel<<<((size_t)5 * 1024 * 64 + 255) / 256, 256, 0, stream>>>(Wed, WedT, 1024, 64, 5);
    convT_kernel<<<((size_t)2560 * 1024 + 255) / 256, 256, 0, stream>>>(Woo, WooT, 2560, 1024, 1);
    convT_kernel<<<((size_t)1024 * 64 + 255) / 256, 256, 0, stream>>>(Wod, WodT, 1024, 64, 1);

    // ---- sequential scan over T ----
    for (int t = 0; t < TT; ++t) {
        // mask state, img_in MLP, build xd=[x|deter], de[:,1024:]=embed
        prep_kernel<<<(BB * 3584) / 256, 256, 0, stream>>>(
            embed, action, isf, Wimg, bimg, stochF, deterF, xd, de, t);

        // GRU pre-activation: [256,2048] x [2048,3072]
        gemm_bf16_wmma<<<dim3(3072 / 64, BB / 128), 256, 0, stream>>>(
            xd, WgT, BB, 3072, 2048, parts, nullptr, nullptr,
            nullptr, 0, 0, 0, 0);

        // LayerNorm + GRU gates -> new deter
        ln_gru_kernel<<<BB, 256, 0, stream>>>(
            parts, ggru, blgru, deterF, deterB, de, out, t);

        // ensemble prior head: xo = elu(deter @ W_eo[idx] + b_eo[idx])
        gemm_bf16_wmma<<<dim3(1024 / 64, BB / 128), 256, 0, stream>>>(
            deterB, WeoT, BB, 1024, 1024, nullptr, xoB, beo,
            eidx, t, (long)1024 * 1024, 1024, 1);
        gemm_bf16_wmma<<<dim3(1, BB / 128), 256, 0, stream>>>(
            xoB, WedT, BB, 64, 1024, edOut, nullptr, bed,
            eidx, t, (long)64 * 1024, 64, 2);
        stats_kernel<<<(BB * SS + 255) / 256, 256, 0, stream>>>(
            edOut, npri, out, nullptr, t, 1120);

        // posterior head: xp = elu([deter|embed] @ W_oo + b_oo)
        gemm_bf16_wmma<<<dim3(1024 / 64, BB / 128), 256, 0, stream>>>(
            de, WooT, BB, 1024, 2560, nullptr, xpB, boo,
            nullptr, 0, 0, 0, 1);
        gemm_bf16_wmma<<<dim3(1, BB / 128), 256, 0, stream>>>(
            xpB, WodT, BB, 64, 1024, odOut, nullptr, bod,
            nullptr, 0, 0, 0, 2);
        stats_kernel<<<(BB * SS + 255) / 256, 256, 0, stream>>>(
            odOut, npost, out, stochF, t, 0);
    }
}
